// InvariantPointAttention_23596550324670
// MI455X (gfx1250) — compile-verified, matrix operand order checked by reading, not run on hardware
//
#include <hip/hip_runtime.h>
#include <hip/hip_bf16.h>
#include <math.h>

typedef __attribute__((ext_vector_type(16))) _Float16 v16h;
typedef __attribute__((ext_vector_type(8)))  _Float16 v8h;
typedef __attribute__((ext_vector_type(8)))  float    v8f;
typedef __attribute__((ext_vector_type(4)))  float    v4f;

#define NRES 512
#define CS   384
#define CZ   128
#define CH   16
#define HH   12
#define PQn  4
#define PVn  8
#define PROJW 1152
#define CATW  2112
#define NN   (NRES * NRES)
#define INFV 100000.0f

// ---------------------------------------------------------------------------
// WMMA helpers (wave32, 16x16x32 f16 -> f32). All operand buffers are laid
// out so that each lane's fragment is two contiguous 16-byte runs.
// ---------------------------------------------------------------------------
__device__ __forceinline__ v8f wmma16(v16h a, v16h b, v8f c) {
  return __builtin_amdgcn_wmma_f32_16x16x32_f16(false, a, false, b, (short)0, c,
                                                false, false);
}

// f16 fragment from row-major [rows x ld] f16 matrix, K-contiguous.
// Lane layout: lanes 0-15 hold K {0..7,16..23}, lanes 16-31 hold {8..15,24..31}.
__device__ __forceinline__ v16h frag16(const _Float16* __restrict__ M, int ld,
                                       int row0, int k0, int lane) {
  const _Float16* p = M + (size_t)(row0 + (lane & 15)) * ld + k0 + ((lane >> 4) << 3);
  v8h lo = *(const v8h*)p;
  v8h hi = *(const v8h*)(p + 16);
  return __builtin_shufflevector(lo, hi, 0, 1, 2, 3, 4, 5, 6, 7,
                                 8, 9, 10, 11, 12, 13, 14, 15);
}

// f32 fragment (A side), contiguous runs, converted to f16.
__device__ __forceinline__ v16h frag32(const float* __restrict__ M, int ld,
                                       int row0, int k0, int lane) {
  const float* p = M + (size_t)(row0 + (lane & 15)) * ld + k0 + ((lane >> 4) << 3);
  const v4f* q = (const v4f*)p;
  v4f x0 = q[0], x1 = q[1], x2 = q[4], x3 = q[5];
  v16h f;
#pragma unroll
  for (int e = 0; e < 4; ++e) {
    f[e]      = (_Float16)x0[e];
    f[e + 4]  = (_Float16)x1[e];
    f[e + 8]  = (_Float16)x2[e];
    f[e + 12] = (_Float16)x3[e];
  }
  return f;
}

// f32 B fragment from row-major [K x ld] (column per lane, K-strided rows).
// Unconditional; lanes are coalesced along n.
__device__ __forceinline__ v16h frag32_bstrided(const float* __restrict__ M, int ld,
                                                int k0, int col0, int lane) {
  const float* p = M + (size_t)(k0 + ((lane >> 4) << 3)) * ld + col0 + (lane & 15);
  v16h f;
#pragma unroll
  for (int e = 0; e < 8; ++e) {
    f[e]     = (_Float16)p[(size_t)e * ld];
    f[e + 8] = (_Float16)p[(size_t)(e + 16) * ld];
  }
  return f;
}

// ---------------------------------------------------------------------------
// Generic GEMM: C[M x N](f32) = A16[M x K] * BT16[N x K]^T + bias
// All of M,N multiples of 16; K multiple of 32. One 16x16 tile per wave.
// ---------------------------------------------------------------------------
__global__ void gemm16_kernel(const _Float16* __restrict__ A, int lda,
                              const _Float16* __restrict__ BT, int ldb,
                              const float* __restrict__ bias,
                              float* __restrict__ C, int ldc,
                              int ntn, int K, int ntiles) {
  int wave = blockIdx.x * (blockDim.x >> 5) + (threadIdx.x >> 5);
  int lane = threadIdx.x & 31;
  if (wave >= ntiles) return;
  int row0 = (wave / ntn) << 4, col0 = (wave % ntn) << 4;
  v8f acc = {};
  for (int k0 = 0; k0 < K; k0 += 32) {
    acc = wmma16(frag16(A, lda, row0, k0, lane),
                 frag16(BT, ldb, col0, k0, lane), acc);
  }
  int n = col0 + (lane & 15);
  float bv = bias[n];
  int rb = row0 + ((lane >> 4) << 3);
#pragma unroll
  for (int r = 0; r < 8; ++r)
    C[(size_t)(rb + r) * ldc + n] = acc[r] + bv;
}

// ---------------------------------------------------------------------------
// Staging kernels (elementwise f32 -> f16 into fragment-friendly layouts)
// ---------------------------------------------------------------------------
__global__ void cvt_s_kernel(const float* __restrict__ s, _Float16* __restrict__ s16) {
  int idx = blockIdx.x * blockDim.x + threadIdx.x;
  if (idx < NRES * CS) s16[idx] = (_Float16)s[idx];
}

__global__ void stage_wcat_kernel(const float* __restrict__ Wq, const float* __restrict__ Wkv,
                                  const float* __restrict__ Wqp, const float* __restrict__ Wkvp,
                                  const float* __restrict__ bq, const float* __restrict__ bkv,
                                  const float* __restrict__ bqp, const float* __restrict__ bkvp,
                                  _Float16* __restrict__ wcatT, float* __restrict__ biascat) {
  int idx = blockIdx.x * blockDim.x + threadIdx.x;   // 1152 * 384
  if (idx >= PROJW * CS) return;
  int n = idx / CS, k = idx % CS;
  float v;
  if (n < 192)      v = Wq[(size_t)k * 192 + n];
  else if (n < 576) v = Wkv[(size_t)k * 384 + (n - 192)];
  else if (n < 720) v = Wqp[(size_t)k * 144 + (n - 576)];
  else              v = Wkvp[(size_t)k * 432 + (n - 720)];
  wcatT[idx] = (_Float16)v;
  if (k == 0) {
    float b = (n < 192) ? bq[n] : (n < 576) ? bkv[n - 192]
            : (n < 720) ? bqp[n - 576] : bkvp[n - 720];
    biascat[n] = b;
  }
}

__global__ void stage_wb_kernel(const float* __restrict__ Wb, _Float16* __restrict__ WbT) {
  int idx = blockIdx.x * blockDim.x + threadIdx.x;   // 16 * 128
  if (idx >= 16 * CZ) return;
  int n = idx / CZ, k = idx % CZ;
  WbT[idx] = (_Float16)((n < HH) ? Wb[(size_t)k * HH + n] : 0.f);
}

__global__ void stage_wout_kernel(const float* __restrict__ Wout, _Float16* __restrict__ WoutT) {
  int idx = blockIdx.x * blockDim.x + threadIdx.x;   // 384 * 2112
  if (idx >= CS * CATW) return;
  int n = idx / CATW, k = idx % CATW;
  WoutT[idx] = (_Float16)Wout[(size_t)k * CS + n];
}

// Packed logit operands: per head h, per row, 32 K-slots = [16 chans | 12 pts | 4 zero]
__global__ void stage_logits_kernel(const float* __restrict__ proj,
                                    const float* __restrict__ qpts,
                                    const float* __restrict__ kpts,
                                    const float* __restrict__ hwv,
                                    _Float16* __restrict__ Aq, _Float16* __restrict__ Bk) {
  int idx = blockIdx.x * blockDim.x + threadIdx.x;   // 2 * 12 * 512 * 32
  if (idx >= 2 * HH * NRES * 32) return;
  int side = idx / (HH * NRES * 32);
  int rem = idx % (HH * NRES * 32);
  int h = rem / (NRES * 32);
  int r2 = rem % (NRES * 32);
  int i = r2 / 32, k = r2 % 32;
  float x = 0.f;
  if (side == 0) {
    const float s1 = rsqrtf(3.f * (float)CH);
    float hw = log1pf(expf(hwv[h])) * sqrtf(1.f / (3.f * (PQn * 9.f / 2.f)));
    if (k < 16)      x = proj[(size_t)i * PROJW + h * CH + k] * s1;
    else if (k < 28) x = qpts[(size_t)i * 144 + h * 12 + (k - 16)] * hw;
    Aq[((size_t)h * NRES + i) * 32 + k] = (_Float16)x;
  } else {
    if (k < 16)      x = proj[(size_t)i * PROJW + 192 + h * 32 + k];
    else if (k < 28) x = kpts[(size_t)i * 144 + h * 12 + (k - 16)];
    Bk[((size_t)h * NRES + i) * 32 + k] = (_Float16)x;
  }
}

// [v(16) | v_pts(24) | pad(8)] transposed per head: bvT[h][n][j]
__global__ void stage_bv_kernel(const float* __restrict__ proj,
                                const float* __restrict__ vpts,
                                _Float16* __restrict__ bvT) {
  int idx = blockIdx.x * blockDim.x + threadIdx.x;   // 12 * 48 * 512
  if (idx >= HH * 48 * NRES) return;
  int h = idx / (48 * NRES);
  int rem = idx % (48 * NRES);
  int n = rem / NRES, j = rem % NRES;
  float x = 0.f;
  if (n < 16)      x = proj[(size_t)j * PROJW + 192 + h * 32 + 16 + n];
  else if (n < 40) x = vpts[(size_t)j * 288 + h * 24 + (n - 16)];
  bvT[idx] = (_Float16)x;
}

// ---------------------------------------------------------------------------
// Frame application + point norms (f32 VALU)
// ---------------------------------------------------------------------------
__global__ void rotate_pts_kernel(const float* __restrict__ proj,
                                  const float* __restrict__ rots,
                                  const float* __restrict__ trans,
                                  float* __restrict__ qpts,
                                  float* __restrict__ kpts,
                                  float* __restrict__ vpts) {
  int idx = blockIdx.x * blockDim.x + threadIdx.x;
  int n = idx / 192, p = idx % 192;
  if (n >= NRES) return;
  const float* R = rots + (size_t)n * 9;
  const float* t = trans + (size_t)n * 3;
  float x, y, z;
  if (p < 48) {
    const float* s = proj + (size_t)n * PROJW + 576;
    x = s[p]; y = s[48 + p]; z = s[96 + p];
  } else {
    int pp = p - 48;
    const float* s = proj + (size_t)n * PROJW + 720;
    x = s[pp]; y = s[144 + pp]; z = s[288 + pp];
  }
  float rx = R[0] * x + R[1] * y + R[2] * z + t[0];
  float ry = R[3] * x + R[4] * y + R[5] * z + t[1];
  float rz = R[6] * x + R[7] * y + R[8] * z + t[2];
  if (p < 48) {
    int h = p >> 2, pq = p & 3;
    float* d = qpts + (size_t)n * 144 + h * 12 + pq * 3;
    d[0] = rx; d[1] = ry; d[2] = rz;
  } else {
    int pp = p - 48;
    int h = pp / 12, pk = pp % 12;
    if (pk < PQn) {
      float* d = kpts + (size_t)n * 144 + h * 12 + pk * 3;
      d[0] = rx; d[1] = ry; d[2] = rz;
    } else {
      float* d = vpts + (size_t)n * 288 + h * 24 + (pk - PQn) * 3;
      d[0] = rx; d[1] = ry; d[2] = rz;
    }
  }
}

__global__ void pts_norm_kernel(const float* __restrict__ qpts,
                                const float* __restrict__ kpts,
                                float* __restrict__ qn, float* __restrict__ kn) {
  int idx = blockIdx.x * blockDim.x + threadIdx.x;
  if (idx >= 2 * NRES * HH) return;
  int which = idx / (NRES * HH);
  int rem = idx % (NRES * HH);
  int n = rem / HH, h = rem % HH;
  const float* s = (which == 0 ? qpts : kpts) + (size_t)n * 144 + h * 12;
  float acc = 0.f;
#pragma unroll
  for (int i = 0; i < 12; ++i) acc += s[i] * s[i];
  (which == 0 ? qn : kn)[n * HH + h] = acc;
}

// ---------------------------------------------------------------------------
// b = z @ Wb + bb -> bws[h][i*N+j]  (z is f32 A operand, contiguous runs)
// ---------------------------------------------------------------------------
__global__ void bproj_kernel(const float* __restrict__ z,
                             const _Float16* __restrict__ WbT,
                             const float* __restrict__ bb,
                             float* __restrict__ bws) {
  int wave = blockIdx.x * (blockDim.x >> 5) + (threadIdx.x >> 5);
  int lane = threadIdx.x & 31;
  int row0 = wave << 4;
  if (row0 >= NN) return;
  v8f acc = {};
#pragma unroll
  for (int k0 = 0; k0 < CZ; k0 += 32) {
    acc = wmma16(frag32(z, CZ, row0, k0, lane),
                 frag16(WbT, CZ, 0, k0, lane), acc);
  }
  int h = lane & 15;
  if (h >= HH) return;
  float bv = bb[h];
  int rb = row0 + ((lane >> 4) << 3);
#pragma unroll
  for (int r = 0; r < 8; ++r)
    bws[(size_t)h * NN + rb + r] = acc[r] + bv;
}

// ---------------------------------------------------------------------------
// Logits: single K=32 WMMA per 16x16 tile; epilogue adds bias/norm/mask terms.
// ---------------------------------------------------------------------------
__global__ void logits_kernel(const _Float16* __restrict__ Aq,
                              const _Float16* __restrict__ Bk,
                              const float* __restrict__ qn,
                              const float* __restrict__ kn,
                              const float* __restrict__ hwv,
                              const float* __restrict__ mask,
                              float* __restrict__ bws) {
  int wave = blockIdx.x * (blockDim.x >> 5) + (threadIdx.x >> 5);
  int lane = threadIdx.x & 31;
  int h = wave >> 10;
  int ti = (wave >> 5) & 31;
  int tj = wave & 31;
  if (h >= HH) return;
  const float s2 = sqrtf(1.f / 3.f);
  float hw = log1pf(expf(hwv[h])) * sqrtf(1.f / (3.f * (PQn * 9.f / 2.f)));
  int i0 = ti << 4, j0 = tj << 4;
  v8f acc = {};
  acc = wmma16(frag16(Aq + (size_t)h * NRES * 32, 32, i0, 0, lane),
               frag16(Bk + (size_t)h * NRES * 32, 32, j0, 0, lane), acc);
  int j = j0 + (lane & 15);
  int rb = i0 + ((lane >> 4) << 3);
  float knj = kn[j * HH + h];
  float mj = mask[j];
#pragma unroll
  for (int r = 0; r < 8; ++r) {
    int i = rb + r;
    size_t idx = (size_t)h * NN + (size_t)i * NRES + j;
    bws[idx] = acc[r] + s2 * bws[idx]
             - 0.5f * hw * (qn[i * HH + h] + knj)
             + INFV * (mask[i] * mj - 1.f);
  }
}

// ---------------------------------------------------------------------------
// Softmax over j; writes probabilities directly as f16 operand buffer
// ---------------------------------------------------------------------------
__global__ void softmax_kernel(const float* __restrict__ bws, _Float16* __restrict__ a16) {
  const float* p = bws + (size_t)blockIdx.x * NRES;
  _Float16* q = a16 + (size_t)blockIdx.x * NRES;
  __shared__ float red[256];
  int t = threadIdx.x;
  float v0 = p[t], v1 = p[t + 256];
  red[t] = fmaxf(v0, v1);
  __syncthreads();
  for (int s = 128; s > 0; s >>= 1) {
    if (t < s) red[t] = fmaxf(red[t], red[t + s]);
    __syncthreads();
  }
  float m = red[0];
  __syncthreads();
  float e0 = __expf(v0 - m), e1 = __expf(v1 - m);
  red[t] = e0 + e1;
  __syncthreads();
  for (int s = 128; s > 0; s >>= 1) {
    if (t < s) red[t] += red[t + s];
    __syncthreads();
  }
  float inv = 1.f / red[0];
  q[t] = (_Float16)(e0 * inv);
  q[t + 256] = (_Float16)(e1 * inv);
}

// ---------------------------------------------------------------------------
// o / o_pt fused: per head, a(512x512) @ [v|v_pts](512x48)
// ---------------------------------------------------------------------------
__global__ void attnout_kernel(const _Float16* __restrict__ a16,
                               const _Float16* __restrict__ bvT,
                               _Float16* __restrict__ cat,
                               float* __restrict__ optw) {
  int wave = blockIdx.x * (blockDim.x >> 5) + (threadIdx.x >> 5);
  int lane = threadIdx.x & 31;
  int h = wave / 96;
  int ti = (wave / 3) % 32;
  int tn = wave % 3;
  if (h >= HH) return;
  int i0 = ti << 4, n0 = tn << 4;
  const _Float16* A = a16 + (size_t)h * NN;
  const _Float16* BT = bvT + (size_t)h * 48 * NRES;
  v8f acc = {};
  for (int k0 = 0; k0 < NRES; k0 += 32)
    acc = wmma16(frag16(A, NRES, i0, k0, lane),
                 frag16(BT, NRES, n0, k0, lane), acc);
  int n = n0 + (lane & 15);
  int rb = i0 + ((lane >> 4) << 3);
#pragma unroll
  for (int r = 0; r < 8; ++r) {
    int i = rb + r;
    if (n < 16) cat[(size_t)i * CATW + h * CH + n] = (_Float16)acc[r];
    else if (n < 40) optw[(size_t)i * 288 + h * 24 + (n - 16)] = acc[r];
  }
}

// ---------------------------------------------------------------------------
// o_pair: per residue i, (16heads x 512) @ z[i](512 x 128); z strided f32 B.
// ---------------------------------------------------------------------------
__global__ void opair_kernel(const _Float16* __restrict__ a16,
                             const float* __restrict__ z,
                             _Float16* __restrict__ cat) {
  int wave = blockIdx.x * (blockDim.x >> 5) + (threadIdx.x >> 5);
  int lane = threadIdx.x & 31;
  int i = wave >> 3;
  int tn = wave & 7;
  if (i >= NRES) return;
  int c0 = tn << 4;
  const _Float16* A = a16 + (size_t)i * NRES;   // rows h, lda = NN
  const float* zr = z + (size_t)i * NRES * CZ;
  v8f acc = {};
  for (int k0 = 0; k0 < NRES; k0 += 32)
    acc = wmma16(frag16(A, NN, 0, k0, lane),
                 frag32_bstrided(zr, CZ, k0, c0, lane), acc);
  int c = c0 + (lane & 15);
  int rb = (lane >> 4) << 3;
#pragma unroll
  for (int r = 0; r < 8; ++r) {
    int h = rb + r;
    if (h < HH) cat[(size_t)i * CATW + 576 + h * CZ + c] = (_Float16)acc[r];
  }
}

// ---------------------------------------------------------------------------
// Inverse-rotate o_pt, norms, scatter into cat (f16)
// ---------------------------------------------------------------------------
__global__ void rotout_kernel(const float* __restrict__ optw,
                              const float* __restrict__ rots,
                              const float* __restrict__ trans,
                              _Float16* __restrict__ cat) {
  int idx = blockIdx.x * blockDim.x + threadIdx.x;
  if (idx >= NRES * HH * PVn) return;
  int n = idx / (HH * PVn);
  int rem = idx % (HH * PVn);
  int h = rem / PVn, p = rem % PVn;
  const float* s = optw + (size_t)n * 288 + h * 24 + p * 3;
  const float* R = rots + (size_t)n * 9;
  const float* t = trans + (size_t)n * 3;
  float x = s[0] - t[0], y = s[1] - t[1], zc = s[2] - t[2];
  float ox = R[0] * x + R[3] * y + R[6] * zc;
  float oy = R[1] * x + R[4] * y + R[7] * zc;
  float oz = R[2] * x + R[5] * y + R[8] * zc;
  float nrm = sqrtf(ox * ox + oy * oy + oz * oz + 1e-8f);
  _Float16* d = cat + (size_t)n * CATW;
  int pp = h * PVn + p;
  d[192 + pp] = (_Float16)ox;
  d[288 + pp] = (_Float16)oy;
  d[384 + pp] = (_Float16)oz;
  d[480 + pp] = (_Float16)nrm;
}

// ---------------------------------------------------------------------------
extern "C" void kernel_launch(void* const* d_in, const int* in_sizes, int n_in,
                              void* d_out, int out_size, void* d_ws, size_t ws_size,
                              hipStream_t stream) {
  const float* s    = (const float*)d_in[0];
  const float* z    = (const float*)d_in[1];
  const float* ttr  = (const float*)d_in[2];
  const float* trot = (const float*)d_in[3];
  const float* mask = (const float*)d_in[4];
  const float* Wq   = (const float*)d_in[5];
  const float* bq   = (const float*)d_in[6];
  const float* Wkv  = (const float*)d_in[7];
  const float* bkv  = (const float*)d_in[8];
  const float* Wqp  = (const float*)d_in[9];
  const float* bqp  = (const float*)d_in[10];
  const float* Wkvp = (const float*)d_in[11];
  const float* bkvp = (const float*)d_in[12];
  const float* Wb   = (const float*)d_in[13];
  const float* bb   = (const float*)d_in[14];
  const float* hwv  = (const float*)d_in[15];
  const float* Wout = (const float*)d_in[16];
  const float* bout = (const float*)d_in[17];
  float* out = (float*)d_out;

  // --- workspace carve-up (f32 region first, then f16 region) ---
  char* base = (char*)d_ws;
  size_t off = 0;
  auto alloc_f32 = [&](size_t n) { float* p = (float*)(base + off); off += n * 4; return p; };
  float* proj    = alloc_f32((size_t)NRES * PROJW);
  float* qpts    = alloc_f32((size_t)NRES * 144);
  float* kpts    = alloc_f32((size_t)NRES * 144);
  float* vpts    = alloc_f32((size_t)NRES * 288);
  float* qn      = alloc_f32((size_t)NRES * HH);
  float* kn      = alloc_f32((size_t)NRES * HH);
  float* bws     = alloc_f32((size_t)HH * NN);
  float* optw    = alloc_f32((size_t)NRES * 288);
  float* biascat = alloc_f32(PROJW);
  off = (off + 255) & ~(size_t)255;
  auto alloc_f16 = [&](size_t n) { _Float16* p = (_Float16*)(base + off); off += n * 2; return p; };
  _Float16* s16    = alloc_f16((size_t)NRES * CS);
  _Float16* wcatT  = alloc_f16((size_t)PROJW * CS);
  _Float16* WbT    = alloc_f16((size_t)16 * CZ);
  _Float16* WoutT  = alloc_f16((size_t)CS * CATW);
  _Float16* Aq     = alloc_f16((size_t)HH * NRES * 32);
  _Float16* Bk     = alloc_f16((size_t)HH * NRES * 32);
  _Float16* bvT    = alloc_f16((size_t)HH * 48 * NRES);
  _Float16* a16    = alloc_f16((size_t)16 * NN);   // 12 valid head rows + 4 pad rows
  _Float16* cat16  = alloc_f16((size_t)NRES * CATW);

  dim3 blk(256);
  auto gblk = [](long n) { return dim3((unsigned)((n + 255) / 256)); };
  auto wblk = [](int waves) { return dim3((unsigned)((waves + 7) / 8)); };

  // --- stage weights / inputs to f16 fragment layouts ---
  cvt_s_kernel<<<gblk((long)NRES * CS), blk, 0, stream>>>(s, s16);
  stage_wcat_kernel<<<gblk((long)PROJW * CS), blk, 0, stream>>>(
      Wq, Wkv, Wqp, Wkvp, bq, bkv, bqp, bkvp, wcatT, biascat);
  stage_wb_kernel<<<gblk(16 * CZ), blk, 0, stream>>>(Wb, WbT);
  stage_wout_kernel<<<gblk((long)CS * CATW), blk, 0, stream>>>(Wout, WoutT);

  // --- fused projection GEMM: proj = s @ [Wq|Wkv|Wqp|Wkvp] + bias ---
  gemm16_kernel<<<wblk(32 * 72), blk, 0, stream>>>(
      s16, CS, wcatT, CS, biascat, proj, PROJW, 72, CS, 32 * 72);

  // --- frame application + norms, then stage attention operands ---
  rotate_pts_kernel<<<gblk((long)NRES * 192), blk, 0, stream>>>(
      proj, trot, ttr, qpts, kpts, vpts);
  pts_norm_kernel<<<gblk(2L * NRES * HH), blk, 0, stream>>>(qpts, kpts, qn, kn);
  stage_logits_kernel<<<gblk(2L * HH * NRES * 32), blk, 0, stream>>>(
      proj, qpts, kpts, hwv, Aq, Bk);
  stage_bv_kernel<<<gblk((long)HH * 48 * NRES), blk, 0, stream>>>(proj, vpts, bvT);

  // --- pair bias (z streamed; resident in 192MB L2 afterwards) ---
  bproj_kernel<<<wblk(NN / 16), blk, 0, stream>>>(z, WbT, bb, bws);

  // --- logits + softmax ---
  logits_kernel<<<wblk(HH * 32 * 32), blk, 0, stream>>>(
      Aq, Bk, qn, kn, hwv, mask, bws);
  softmax_kernel<<<dim3(HH * NRES), blk, 0, stream>>>(bws, a16);

  // --- attention outputs ---
  attnout_kernel<<<wblk(HH * 32 * 3), blk, 0, stream>>>(a16, bvT, cat16, optw);
  opair_kernel<<<wblk(NRES * 8), blk, 0, stream>>>(a16, z, cat16);
  rotout_kernel<<<gblk((long)NRES * HH * PVn), blk, 0, stream>>>(
      optw, trot, ttr, cat16);

  // --- final projection ---
  gemm16_kernel<<<wblk(32 * 24), blk, 0, stream>>>(
      cat16, CATW, WoutT, CATW, bout, out, CS, 24, CATW, 32 * 24);
}
